// MultiHeadAttention_16922171146418
// MI455X (gfx1250) — matrix operand order repaired and checked by reading, not compiled, not run on hardware
//
#include <hip/hip_runtime.h>

// Problem constants (match the JAX reference)
#define BB  4
#define TT  2048
#define CC  1024
#define HH  16
#define DHD 64

typedef __attribute__((ext_vector_type(16))) __bf16 v16bf;
typedef __attribute__((ext_vector_type(8)))  __bf16 v8bf;
typedef __attribute__((ext_vector_type(8)))  float  v8f;

union V16U { v16bf v; v8bf h[2]; };

// A-matrix fragment (16x32 bf16, M x K), source row-major [M][K] with leading dim ld.
// Lane l: row = l%16; K chunks = hb*8 + {0..7} and 16 + hb*8 + {0..7} (hb = l>=16).
__device__ __forceinline__ v16bf load_a_frag(const __bf16* base, int ld, int rbase,
                                             int kb, int ln, int hf) {
  const __bf16* p = base + (size_t)(rbase + ln) * ld + kb + hf * 8;
  V16U u;
  u.h[0] = *(const v8bf*)p;
  u.h[1] = *(const v8bf*)(p + 16);
  return u.v;
}

// B-matrix fragment (32x16 bf16, K x N), source stored as [N][K] row-major (i.e. B columns
// are contiguous rows of the source). Lane l: col = l%16; K chunks = hb*8 + {0..7} and 16 + hb*8 + {0..7}.
__device__ __forceinline__ v16bf load_b_frag(const __bf16* base, int ld, int nbase,
                                             int kb, int ln, int hf) {
  const __bf16* p = base + (size_t)(nbase + ln) * ld + kb + hf * 8;
  V16U u;
  u.h[0] = *(const v8bf*)p;
  u.h[1] = *(const v8bf*)(p + 16);
  return u.v;
}

__device__ __forceinline__ v8f wmma_bf16(v16bf a, v16bf b, v8f c) {
  return __builtin_amdgcn_wmma_f32_16x16x32_bf16(false, a, false, b, (short)0, c,
                                                 false, false);
}

// ---------------- Kernel 0a: fp32 -> bf16 elementwise ----------------
__global__ void cvt_kernel(const float* __restrict__ src, __bf16* __restrict__ dst, int n) {
  int i = blockIdx.x * blockDim.x + threadIdx.x;
  if (i < n) dst[i] = (__bf16)src[i];
}

// ---------------- Kernel 0b: Wq/Wk/Wv [H][C][DH] fp32 -> [H][DH][C] bf16 ----------------
__global__ void wtr_kernel(const float* __restrict__ src, __bf16* __restrict__ dst) {
  int i = blockIdx.x * blockDim.x + threadIdx.x;
  if (i >= HH * CC * DHD) return;
  int h   = i >> 16;        // C*DH = 65536
  int rem = i & 65535;
  int c   = rem >> 6;       // DH = 64
  int d   = rem & 63;
  dst[(h << 16) + d * CC + c] = (__bf16)src[i];
}

// ---------------- Kernel 1: QKV projections ----------------
// grid = (B*H, T/64), block = 192 (6 waves). wave -> (rowgroup in {0,1}) x (mat in {q,k,v}).
// Each wave: 2 M-tiles x 4 N-tiles, double-buffered K loop (8 WMMA / 12 loads per K-step).
// q stored pre-scaled by DH^-0.5. q,k: [B,H,T,DH] bf16; v: transposed [B,H,DH,T] bf16.
__global__ void __launch_bounds__(192) qkv_kernel(const __bf16* __restrict__ xb,
                                                  const __bf16* __restrict__ WqT,
                                                  const __bf16* __restrict__ WkT,
                                                  const __bf16* __restrict__ WvT,
                                                  __bf16* __restrict__ q,
                                                  __bf16* __restrict__ k,
                                                  __bf16* __restrict__ vT) {
  int tid = threadIdx.x;
  int lane = tid & 31, ln = lane & 15, hf = lane >> 4;
  int wave = tid >> 5;
  int mt = wave & 1, mat = wave >> 1;
  int b = blockIdx.x / HH, h = blockIdx.x % HH;
  int r0 = blockIdx.y * 64 + mt * 32;
  int r1 = r0 + 16;

  const __bf16* xa = xb + (size_t)b * TT * CC;
  const __bf16* WT = (mat == 0 ? WqT : (mat == 1 ? WkT : WvT)) + (size_t)h * DHD * CC;

  v8f acc[2][4] = {};
  v16bf a0[2], b0[4], a1[2], b1[4];

  a0[0] = load_a_frag(xa, CC, r0, 0, ln, hf);
  a0[1] = load_a_frag(xa, CC, r1, 0, ln, hf);
#pragma unroll
  for (int nt = 0; nt < 4; ++nt) b0[nt] = load_b_frag(WT, CC, nt * 16, 0, ln, hf);

  for (int kb = 0; kb < CC; kb += 64) {
    int k1 = kb + 32;
    a1[0] = load_a_frag(xa, CC, r0, k1, ln, hf);
    a1[1] = load_a_frag(xa, CC, r1, k1, ln, hf);
#pragma unroll
    for (int nt = 0; nt < 4; ++nt) b1[nt] = load_b_frag(WT, CC, nt * 16, k1, ln, hf);
#pragma unroll
    for (int nt = 0; nt < 4; ++nt) {
      acc[0][nt] = wmma_bf16(a0[0], b0[nt], acc[0][nt]);
      acc[1][nt] = wmma_bf16(a0[1], b0[nt], acc[1][nt]);
    }
    int k2 = (kb + 64 < CC) ? (kb + 64) : 0;  // clamp: valid-but-unused prefetch on tail
    a0[0] = load_a_frag(xa, CC, r0, k2, ln, hf);
    a0[1] = load_a_frag(xa, CC, r1, k2, ln, hf);
#pragma unroll
    for (int nt = 0; nt < 4; ++nt) b0[nt] = load_b_frag(WT, CC, nt * 16, k2, ln, hf);
#pragma unroll
    for (int nt = 0; nt < 4; ++nt) {
      acc[0][nt] = wmma_bf16(a1[0], b1[nt], acc[0][nt]);
      acc[1][nt] = wmma_bf16(a1[1], b1[nt], acc[1][nt]);
    }
  }

  size_t bh = (size_t)(b * HH + h);
  float sc = (mat == 0) ? 0.125f : 1.0f;  // fold DH^-0.5 into q (exact in bf16)
  if (mat < 2) {
    __bf16* dst = (mat == 0 ? q : k) + bh * TT * DHD;
#pragma unroll
    for (int g = 0; g < 2; ++g)
#pragma unroll
      for (int nt = 0; nt < 4; ++nt)
#pragma unroll
        for (int j = 0; j < 8; ++j) {
          int row = (g ? r1 : r0) + j + 8 * hf;
          int col = nt * 16 + ln;
          dst[(size_t)row * DHD + col] = (__bf16)(acc[g][nt][j] * sc);
        }
  } else {
    __bf16* dst = vT + bh * DHD * TT;  // [DH][T]
#pragma unroll
    for (int g = 0; g < 2; ++g)
#pragma unroll
      for (int nt = 0; nt < 4; ++nt)
#pragma unroll
        for (int j = 0; j < 8; ++j) {
          int t = (g ? r1 : r0) + j + 8 * hf;
          int d = nt * 16 + ln;
          dst[(size_t)d * TT + t] = (__bf16)acc[g][nt][j];
        }
  }
}

// ---------------- Kernel 2: causal flash attention ----------------
// grid = B*H*(T/128), block = 256 (8 waves). Each wave owns one 16-row q tile and
// streams 32-key blocks with online softmax. All K/V fragment loads for a block are
// batched up-front so V loads overlap the softmax VALU work. P is restaged through
// per-wave LDS to convert WMMA C-layout -> A-layout.
__global__ void __launch_bounds__(256) flash_kernel(const __bf16* __restrict__ q,
                                                    const __bf16* __restrict__ k,
                                                    const __bf16* __restrict__ vT,
                                                    __bf16* __restrict__ att) {
  __shared__ __bf16 pbuf[8][16][32];  // 8 KB, one 16x32 P tile per wave

  int tid = threadIdx.x, lane = tid & 31, ln = lane & 15, hf = lane >> 4, wave = tid >> 5;
  int bh = blockIdx.x >> 4, qc = blockIdx.x & 15;
  int b = bh / HH, h = bh % HH;
  int qb = qc * 128 + wave * 16;

  const __bf16* qp = q  + (size_t)bh * TT * DHD;
  const __bf16* kp = k  + (size_t)bh * TT * DHD;
  const __bf16* vp = vT + (size_t)bh * DHD * TT;

  // Q fragments (K = DH = 64 -> two 16x32 A-frags), reused across all key blocks.
  // q was pre-scaled by DH^-0.5 in the projection kernel.
  v16bf aq0 = load_a_frag(qp, DHD, qb, 0,  ln, hf);
  v16bf aq1 = load_a_frag(qp, DHD, qb, 32, ln, hf);

  v8f o[4] = {};
  float m[8], l[8];
#pragma unroll
  for (int j = 0; j < 8; ++j) { m[j] = -1e30f; l[j] = 0.f; }

  for (int sb = 0; sb < qb + 16; sb += 32) {
    // Batch-issue all fragment loads for this key block.
    v16bf kf[4], vf[4];
    kf[0] = load_b_frag(kp, DHD, sb,      0,  ln, hf);
    kf[1] = load_b_frag(kp, DHD, sb,      32, ln, hf);
    kf[2] = load_b_frag(kp, DHD, sb + 16, 0,  ln, hf);
    kf[3] = load_b_frag(kp, DHD, sb + 16, 32, ln, hf);
#pragma unroll
    for (int nt = 0; nt < 4; ++nt)
      vf[nt] = load_b_frag(vp, TT, nt * 16, sb, ln, hf);  // vT rows = head dims

    v8f s0 = {}, s1 = {};
    s0 = wmma_bf16(aq0, kf[0], s0);
    s0 = wmma_bf16(aq1, kf[1], s0);
    s1 = wmma_bf16(aq0, kf[2], s1);
    s1 = wmma_bf16(aq1, kf[3], s1);

    if (sb + 31 > qb) {  // diagonal block: apply causal mask (col > row -> -inf)
      int c0 = sb + ln, c1 = sb + 16 + ln;
#pragma unroll
      for (int j = 0; j < 8; ++j) {
        int row = qb + j + 8 * hf;
        if (c0 > row) s0[j] = -1e30f;
        if (c1 > row) s1[j] = -1e30f;
      }
    }

    float alpha[8];
    v8f p0, p1;
#pragma unroll
    for (int j = 0; j < 8; ++j) {
      // row max over 32 key columns (16-lane group reduction in C-layout)
      float t = fmaxf(s0[j], s1[j]);
      t = fmaxf(t, __shfl_xor(t, 1, 32));
      t = fmaxf(t, __shfl_xor(t, 2, 32));
      t = fmaxf(t, __shfl_xor(t, 4, 32));
      t = fmaxf(t, __shfl_xor(t, 8, 32));
      float mn = fmaxf(m[j], t);
      alpha[j] = __expf(m[j] - mn);
      m[j] = mn;
      p0[j] = __expf(s0[j] - mn);
      p1[j] = __expf(s1[j] - mn);
      float rs = p0[j] + p1[j];
      rs += __shfl_xor(rs, 1, 32);
      rs += __shfl_xor(rs, 2, 32);
      rs += __shfl_xor(rs, 4, 32);
      rs += __shfl_xor(rs, 8, 32);
      l[j] = l[j] * alpha[j] + rs;
    }

#pragma unroll
    for (int nt = 0; nt < 4; ++nt)
#pragma unroll
      for (int j = 0; j < 8; ++j) o[nt][j] *= alpha[j];

    // Stage P (C-layout) into LDS as a row-major 16x32 bf16 tile.
#pragma unroll
    for (int j = 0; j < 8; ++j) {
      pbuf[wave][j + 8 * hf][ln]      = (__bf16)p0[j];
      pbuf[wave][j + 8 * hf][16 + ln] = (__bf16)p1[j];
    }
    // Same-wave DS ops are in-order; explicit wait before re-reading the tile.
    asm volatile("s_wait_dscnt 0x0" ::: "memory");

    // Re-read P in A-layout.
    V16U up;
    up.h[0] = *(const v8bf*)&pbuf[wave][ln][hf * 8];
    up.h[1] = *(const v8bf*)&pbuf[wave][ln][16 + hf * 8];
    v16bf pa = up.v;

#pragma unroll
    for (int nt = 0; nt < 4; ++nt)
      o[nt] = wmma_bf16(pa, vf[nt], o[nt]);
  }

  // Normalize and store concat-head output att[B][T][C] (bf16, consumed by proj GEMM).
  __bf16* op = att + (size_t)b * TT * CC + h * DHD;
#pragma unroll
  for (int j = 0; j < 8; ++j) {
    float inv = 1.0f / l[j];
    int row = qb + j + 8 * hf;
#pragma unroll
    for (int nt = 0; nt < 4; ++nt)
      op[(size_t)row * CC + nt * 16 + ln] = (__bf16)(o[nt][j] * inv);
  }
}

// ---------------- Kernel 3: output projection y = att @ Wo^T + bo (fp32 out) ----------------
// grid = (M/64, N/256), block = 256 (8 waves = 2 row-waves x 4 col-waves).
// Each wave: 2 M-tiles x 4 N-tiles, double-buffered K loop.
__global__ void __launch_bounds__(256) proj_kernel(const __bf16* __restrict__ att,
                                                   const __bf16* __restrict__ Wo,
                                                   const float* __restrict__ bo,
                                                   float* __restrict__ out) {
  int tid = threadIdx.x, lane = tid & 31, ln = lane & 15, hf = lane >> 4, wave = tid >> 5;
  int mt = wave & 1, cg = wave >> 1;
  int r0 = blockIdx.x * 64 + mt * 32;
  int r1 = r0 + 16;
  int nbase = blockIdx.y * 256 + cg * 64;

  v8f acc[2][4] = {};
  v16bf a0[2], b0[4], a1[2], b1[4];

  a0[0] = load_a_frag(att, CC, r0, 0, ln, hf);
  a0[1] = load_a_frag(att, CC, r1, 0, ln, hf);
#pragma unroll
  for (int nt = 0; nt < 4; ++nt) b0[nt] = load_b_frag(Wo, CC, nbase + nt * 16, 0, ln, hf);

  for (int kb = 0; kb < CC; kb += 64) {
    int k1 = kb + 32;
    a1[0] = load_a_frag(att, CC, r0, k1, ln, hf);
    a1[1] = load_a_frag(att, CC, r1, k1, ln, hf);
#pragma unroll
    for (int nt = 0; nt < 4; ++nt) b1[nt] = load_b_frag(Wo, CC, nbase + nt * 16, k1, ln, hf);
#pragma unroll
    for (int nt = 0; nt < 4; ++nt) {
      acc[0][nt] = wmma_bf16(a0[0], b0[nt], acc[0][nt]);
      acc[1][nt] = wmma_bf16(a0[1], b0[nt], acc[1][nt]);
    }
    int k2 = (kb + 64 < CC) ? (kb + 64) : 0;
    a0[0] = load_a_frag(att, CC, r0, k2, ln, hf);
    a0[1] = load_a_frag(att, CC, r1, k2, ln, hf);
#pragma unroll
    for (int nt = 0; nt < 4; ++nt) b0[nt] = load_b_frag(Wo, CC, nbase + nt * 16, k2, ln, hf);
#pragma unroll
    for (int nt = 0; nt < 4; ++nt) {
      acc[0][nt] = wmma_bf16(a1[0], b1[nt], acc[0][nt]);
      acc[1][nt] = wmma_bf16(a1[1], b1[nt], acc[1][nt]);
    }
  }

#pragma unroll
  for (int g = 0; g < 2; ++g)
#pragma unroll
    for (int nt = 0; nt < 4; ++nt) {
      int col = nbase + nt * 16 + ln;
      float bias = bo[col];
#pragma unroll
      for (int j = 0; j < 8; ++j) {
        int row = (g ? r1 : r0) + j + 8 * hf;
        out[(size_t)row * CC + col] = acc[g][nt][j] + bias;
      }
    }
}

// ---------------- Host launcher ----------------
extern "C" void kernel_launch(void* const* d_in, const int* in_sizes, int n_in,
                              void* d_out, int out_size, void* d_ws, size_t ws_size,
                              hipStream_t stream) {
  const float* x  = (const float*)d_in[0];
  const float* Wq = (const float*)d_in[1];
  const float* Wk = (const float*)d_in[2];
  const float* Wv = (const float*)d_in[3];
  const float* Wo = (const float*)d_in[4];
  const float* bo = (const float*)d_in[5];
  float* out = (float*)d_out;

  char* w = (char*)d_ws;
  size_t off = 0;
  auto alloc = [&](size_t bytes) -> void* {
    void* p = w + off;
    off = (off + bytes + 255) & ~(size_t)255;
    return p;
  };

  const size_t nx  = (size_t)BB * TT * CC;        // 8M elems
  const size_t nw  = (size_t)HH * CC * DHD;       // 1M elems
  const size_t nqk = (size_t)BB * HH * TT * DHD;  // 8M elems

  __bf16* xb   = (__bf16*)alloc(nx * 2);
  __bf16* WqT  = (__bf16*)alloc(nw * 2);
  __bf16* WkT  = (__bf16*)alloc(nw * 2);
  __bf16* WvT  = (__bf16*)alloc(nw * 2);
  __bf16* Wob  = (__bf16*)alloc((size_t)CC * CC * 2);
  __bf16* qb   = (__bf16*)alloc(nqk * 2);
  __bf16* kb   = (__bf16*)alloc(nqk * 2);
  __bf16* vTb  = (__bf16*)alloc(nqk * 2);
  __bf16* attb = (__bf16*)alloc(nx * 2);
  (void)ws_size; (void)in_sizes; (void)n_in; (void)out_size;

  // 0) fp32 -> bf16 conversions / weight transposes
  cvt_kernel<<<(unsigned)((nx + 255) / 256), 256, 0, stream>>>(x, xb, (int)nx);
  cvt_kernel<<<(unsigned)((CC * CC + 255) / 256), 256, 0, stream>>>(Wo, Wob, CC * CC);
  wtr_kernel<<<(unsigned)((nw + 255) / 256), 256, 0, stream>>>(Wq, WqT);
  wtr_kernel<<<(unsigned)((nw + 255) / 256), 256, 0, stream>>>(Wk, WkT);
  wtr_kernel<<<(unsigned)((nw + 255) / 256), 256, 0, stream>>>(Wv, WvT);

  // 1) QKV projections (q pre-scaled by DH^-0.5)
  qkv_kernel<<<dim3(BB * HH, TT / 64), 192, 0, stream>>>(xb, WqT, WkT, WvT, qb, kb, vTb);

  // 2) causal flash attention
  flash_kernel<<<BB * HH * (TT / 128), 256, 0, stream>>>(qb, kb, vTb, attb);

  // 3) output projection + bias (fp32 result)
  proj_kernel<<<dim3((BB * TT) / 64, CC / 256), 256, 0, stream>>>(attb, Wob, bo, out);
}
